// GraphTransformer_17111149707850
// MI455X (gfx1250) — matrix operand order repaired and checked by reading, not compiled, not run on hardware
//
#include <hip/hip_runtime.h>
#include <math.h>

// ---------------------------------------------------------------------------
// GraphTransformer attention (segment softmax over edges) for gfx1250.
// Memory-bound (~0.8 GB traffic, ~35us at 23.3 TB/s); scores computed on the
// matrix pipe via v_wmma_f32_16x16x32_bf16 (diagonal of Q_tile x Ke_tile^T).
// ---------------------------------------------------------------------------

typedef __bf16 v16bf __attribute__((ext_vector_type(16)));
typedef __bf16 v8bf  __attribute__((ext_vector_type(8)));
typedef float  v8f   __attribute__((ext_vector_type(8)));

#define GT_H 16
#define GT_C 32

__device__ __forceinline__ unsigned short f2bf_bits(float f) {
    unsigned u = __float_as_uint(f);
    u += 0x7FFFu + ((u >> 16) & 1u);          // round-to-nearest-even
    return (unsigned short)(u >> 16);
}

__device__ __forceinline__ __bf16 f2bf(float f) {
    unsigned short hs = f2bf_bits(f);
    __bf16 r;
    __builtin_memcpy(&r, &hs, sizeof(r));
    return r;
}

// Sign-aware float atomic max (buffer pre-initialized to -inf).
__device__ __forceinline__ void atomicMaxFloat(float* addr, float val) {
    if (val >= 0.0f) atomicMax((int*)addr, __float_as_int(val));
    else             atomicMin((unsigned int*)addr, __float_as_uint(val));
}

// ---------------------------------------------------------------------------
// Kernel 0: init out=0, m=-inf, denom=0
// ---------------------------------------------------------------------------
__global__ void gt_init(float* __restrict__ out, float* __restrict__ m,
                        float* __restrict__ denom, long nOut, long nRows) {
    long i = (long)blockIdx.x * blockDim.x + threadIdx.x;
    if (i < nOut) out[i] = 0.0f;
    if (i < nRows) { m[i] = -INFINITY; denom[i] = 0.0f; }
}

// ---------------------------------------------------------------------------
// Kernel 1: fused RMSNorm of q (store bf16 for WMMA A) and k (store f32).
// One wave32 per (n,h) row; lane == channel.
// ---------------------------------------------------------------------------
__global__ void gt_rmsnorm(const float* __restrict__ q, const float* __restrict__ k,
                           const float* __restrict__ wq, const float* __restrict__ wk,
                           unsigned short* __restrict__ qnb, float* __restrict__ kn,
                           long rows) {
    int  lane = threadIdx.x & 31;
    long row  = (long)blockIdx.x * (blockDim.x >> 5) + (threadIdx.x >> 5);
    if (row >= rows) return;
    size_t base = (size_t)row * GT_C + lane;
    float xq = q[base], xk = k[base];
    float sq = xq * xq, sk = xk * xk;
#pragma unroll
    for (int off = 16; off >= 1; off >>= 1) {
        sq += __shfl_xor(sq, off, 32);
        sk += __shfl_xor(sk, off, 32);
    }
    float invq = rsqrtf(sq * (1.0f / GT_C) + 1e-5f);
    float invk = rsqrtf(sk * (1.0f / GT_C) + 1e-5f);
    qnb[base] = f2bf_bits(xq * invq * wq[lane]);
    kn[base]  = xk * invk * wk[lane];
}

// ---------------------------------------------------------------------------
// Kernel 2: edge scores via WMMA. One wave per 16-edge tile; per head h:
//   D(16x16) = Qn_tile(16x32,bf16) x KeT(32x16,bf16), s[e][h] = diag(D)*scale.
// Fragments built per-lane straight from global memory, contiguous k order in A and B:
//   A (16-bit, 16x32): lanes 0-15 row=lane, K 0..15;
//                      lanes 16-31 row=lane-16, K 16..31.
//   B (16-bit, 32x16): lanes 0-15 col=lane, K 0..15; lanes 16-31 K 16..31.
// Diagonal of D: edge r -> (VGPR r, lane r); edge 8+r -> (VGPR r, lane 24+r);
// those lanes already own that edge's dst/eIdx.
// ---------------------------------------------------------------------------
__global__ void __launch_bounds__(128)
gt_scores_wmma(const unsigned short* __restrict__ qnb_raw,
               const float* __restrict__ kn, const float* __restrict__ eF,
               const int* __restrict__ esrc, const int* __restrict__ edst,
               float* __restrict__ sbuf, float* __restrict__ m, long NE) {
    const __bf16* qnb = (const __bf16*)qnb_raw;
    int  lane   = threadIdx.x & 31;
    long nTiles = (NE + 15) >> 4;
    long tile   = (long)blockIdx.x * (blockDim.x >> 5) + (threadIdx.x >> 5);
    if (tile >= nTiles) return;             // wave-uniform: EXEC stays full

    int  eLocal = lane & 15;
    long eIdx   = tile * 16 + eLocal;
    bool valid  = (eIdx < NE);
    if (!valid) eIdx = NE - 1;              // clamp loads; suppress stores only
    int src = esrc[eIdx];
    int dst = edst[eIdx];

    const int  cloA = (lane < 16) ? 0 : 16;  // A fragment K-range
    const int  cbB  = (lane < 16) ? 0 : 16;  // B fragment K-range
    const bool isW  = (lane < 8) || (lane >= 24);
    const float qk_scale = 0.1767766952966369f; // 1/sqrt(32)

    for (int h = 0; h < GT_H; ++h) {
        // ---- A fragment: qn[dst][h][*] (bf16, already normalized) ----
        const __bf16* arow = qnb + ((size_t)dst * GT_H + h) * GT_C;
        v8bf a0 = *(const v8bf*)(arow + cloA);
        v8bf a1 = *(const v8bf*)(arow + cloA + 8);
        v16bf A;
#pragma unroll
        for (int i = 0; i < 8; ++i) { A[i] = a0[i]; A[8 + i] = a1[i]; }

        // ---- B fragment: ke = kn[src][h][*] + e[edge][h][*]  (f32 add -> bf16)
        const float* kp = kn + ((size_t)src  * GT_H + h) * GT_C + cbB;
        const float* ep = eF + ((size_t)eIdx * GT_H + h) * GT_C + cbB;
        v16bf B;
#pragma unroll
        for (int i = 0; i < 16; i += 4) {
            float4 kv = *(const float4*)(kp + i);
            float4 ev = *(const float4*)(ep + i);
            B[i + 0] = f2bf(kv.x + ev.x);
            B[i + 1] = f2bf(kv.y + ev.y);
            B[i + 2] = f2bf(kv.z + ev.z);
            B[i + 3] = f2bf(kv.w + ev.w);
        }

        v8f Cz = {0.f, 0.f, 0.f, 0.f, 0.f, 0.f, 0.f, 0.f};
        v8f D = __builtin_amdgcn_wmma_f32_16x16x32_bf16(
            /*neg_a=*/false, A, /*neg_b=*/false, B,
            /*c_mod=*/(short)0, Cz, /*reuse_a=*/false, /*reuse_b=*/false);

        // ---- extract diagonal ----
        float sval = 0.0f;
#pragma unroll
        for (int r = 0; r < 8; ++r)
            if (lane == r || lane == 24 + r) sval = D[r];

        if (isW && valid) {
            sval *= qk_scale;
            sbuf[(size_t)eIdx * GT_H + h] = sval;
            atomicMaxFloat(m + (size_t)dst * GT_H + h, sval);
        }
    }
}

// ---------------------------------------------------------------------------
// Kernel 3: alpha = exp(s - m[dst]); scatter denom and num (into d_out).
// One lane per (edge, head, channel): coalesced v/e reads, L2-resident atomics.
// ---------------------------------------------------------------------------
__global__ void gt_scatter(const float* __restrict__ sbuf, const float* __restrict__ m,
                           const float* __restrict__ v, const float* __restrict__ eF,
                           const int* __restrict__ esrc, const int* __restrict__ edst,
                           float* __restrict__ out, float* __restrict__ denom, long NE) {
    long t    = (long)blockIdx.x * blockDim.x + threadIdx.x;
    long pair = t >> 5;
    int  c    = (int)(t & 31);
    if (pair >= NE * GT_H) return;
    long edge = pair >> 4;
    int  h    = (int)(pair & 15);
    int  dst  = edst[edge];
    int  src  = esrc[edge];

    float alpha = expf(sbuf[pair] - m[(size_t)dst * GT_H + h]);
    if (c == 0) atomicAdd(denom + (size_t)dst * GT_H + h, alpha);

    float ve = v[((size_t)src * GT_H + h) * GT_C + c]
             + eF[((size_t)edge * GT_H + h) * GT_C + c];
    atomicAdd(out + ((size_t)dst * GT_H + h) * GT_C + c, alpha * ve);
}

// ---------------------------------------------------------------------------
// Kernel 4: out = num / (denom > 0 ? denom : 1)   (empty segments -> 0)
// ---------------------------------------------------------------------------
__global__ void gt_finalize(float* __restrict__ out, const float* __restrict__ denom,
                            long n) {
    long i = (long)blockIdx.x * blockDim.x + threadIdx.x;
    if (i >= n) return;
    float d = denom[i >> 5];
    out[i] = out[i] / ((d > 0.0f) ? d : 1.0f);
}

// ---------------------------------------------------------------------------
extern "C" void kernel_launch(void* const* d_in, const int* in_sizes, int n_in,
                              void* d_out, int out_size, void* d_ws, size_t ws_size,
                              hipStream_t stream) {
    const float* q   = (const float*)d_in[0];
    const float* k   = (const float*)d_in[1];
    const float* v   = (const float*)d_in[2];
    const float* eF  = (const float*)d_in[3];
    const float* wq  = (const float*)d_in[4];
    const float* wk  = (const float*)d_in[5];
    const int*   es  = (const int*)d_in[6];
    const int*   ed  = (const int*)d_in[7];
    float*       out = (float*)d_out;

    const long N    = (long)in_sizes[0] / (GT_H * GT_C);
    const long NE   = (long)in_sizes[6];
    const long rows = N * GT_H;          // (node, head) rows
    const long nOut = rows * GT_C;

    // Workspace carve-out (256B aligned regions): ~64 MB total.
    char* p = (char*)d_ws;
    auto carve = [&](size_t bytes) {
        char* r = p;
        p += (bytes + 255) & ~(size_t)255;
        return r;
    };
    unsigned short* qnb   = (unsigned short*)carve((size_t)rows * GT_C * sizeof(unsigned short));
    float*          kn    = (float*)carve((size_t)rows * GT_C * sizeof(float));
    float*          sbuf  = (float*)carve((size_t)NE * GT_H * sizeof(float));
    float*          m     = (float*)carve((size_t)rows * sizeof(float));
    float*          denom = (float*)carve((size_t)rows * sizeof(float));

    // 0) init
    {
        long nthr = nOut;
        gt_init<<<(unsigned)((nthr + 255) / 256), 256, 0, stream>>>(out, m, denom, nOut, rows);
    }
    // 1) RMSNorm q->bf16, k->f32 (wave per row, 8 rows per 256-thread block)
    {
        long nblk = (rows + 7) / 8;
        gt_rmsnorm<<<(unsigned)nblk, 256, 0, stream>>>(q, k, wq, wk, qnb, kn, rows);
    }
    // 2) WMMA edge scores + segment max (wave per 16-edge tile, 4 waves/block)
    {
        long nTiles = (NE + 15) >> 4;
        long nblk   = (nTiles + 3) / 4;
        gt_scores_wmma<<<(unsigned)nblk, 128, 0, stream>>>(qnb, kn, eF, es, ed, sbuf, m, NE);
    }
    // 3) softmax scatter (lane per (edge, head, channel))
    {
        long nthr = NE * GT_H * GT_C;
        gt_scatter<<<(unsigned)((nthr + 255) / 256), 256, 0, stream>>>(
            sbuf, m, v, eF, es, ed, out, denom, NE);
    }
    // 4) divide by denom
    gt_finalize<<<(unsigned)((nOut + 255) / 256), 256, 0, stream>>>(out, denom, nOut);
}
